// StereoINR_34643206209738
// MI455X (gfx1250) — compile-verified, matrix operand order checked by reading, not run on hardware
//
#include <hip/hip_runtime.h>
#include <hip/hip_bf16.h>
#include <cstdint>
#include <cstddef>

#define DI __device__ __forceinline__

typedef __attribute__((ext_vector_type(16))) __bf16    v16bf;
typedef __attribute__((ext_vector_type(8)))  float     v8f;
typedef __attribute__((ext_vector_type(4)))  unsigned  u32x4;

// ---------------- problem constants ----------------
constexpr int N_   = 2;
constexpr int CF   = 64;
constexpr int H_   = 64;
constexpr int W_   = 192;
constexpr int HW   = H_ * W_;
constexpr int Q_   = 98304;
constexpr int HID  = 128;
constexpr int QT   = 32;           // queries per workgroup
constexpr int ROWS = QT * 4;       // 128 MLP rows per workgroup
constexpr int STRIDE = 264;        // LDS activation row stride in bf16 elems (256 + pad)

// ---------------- workspace layout (bytes) ----------------
constexpr size_t SZ_CF     = (size_t)N_ * HID * HW * 2;  // one bf16 tensor [N,128,H,W]
constexpr size_t OFF_COEF0 = 0;
constexpr size_t OFF_COEF1 = SZ_CF;
constexpr size_t OFF_FREQ0 = 2 * SZ_CF;
constexpr size_t OFF_FREQ1 = 3 * SZ_CF;
constexpr size_t OFF_WPK   = 4 * SZ_CF;                  // packed weights, 233472 bf16
constexpr size_t OFF_BPK   = OFF_WPK + (size_t)233472 * 2;  // 1040 f32 biases
// total ~25.6 MB

// packed-weight flat element offsets per layer
constexpr int WO1 = 0;        // 128->256 : 16 nt * 4 kt * 512
constexpr int WO2 = 32768;    // 256->256 : 16 nt * 8 kt * 512
constexpr int WO3 = 98304;
constexpr int WO4 = 163840;
constexpr int WO5 = 229376;   // 256->16(pad) : 1 nt * 8 kt * 512
constexpr int WTOT = 233472;

// K index inside a 16-bit A/B fragment: V>=4 selects upper K half, lane group adds 8, pair V, half h
DI int frag_kbase(int V, int lane) {
    int kb = (V < 4) ? (2 * V) : (16 + 2 * (V - 4));
    return kb + 8 * (lane >> 4);
}

// ================= conv 3x3 (feat -> coef/freq, bf16) =================
__global__ __launch_bounds__(256) void conv3x3_kernel(
    const float* __restrict__ feat_l, const float* __restrict__ feat_r,
    const float* __restrict__ coef_w, const float* __restrict__ coef_b,
    const float* __restrict__ freq_w, const float* __restrict__ freq_b,
    __bf16* __restrict__ coefL, __bf16* __restrict__ coefR,
    __bf16* __restrict__ freqL, __bf16* __restrict__ freqR)
{
    __shared__ float wl[CF * 9];
    __shared__ float wf[CF * 9];
    const int tid  = threadIdx.x;
    const int co   = blockIdx.y;
    const int z    = blockIdx.z;
    const int side = z >> 1;
    const int n    = z & 1;
    const float* feat = side ? feat_r : feat_l;
    for (int i = tid; i < CF * 9; i += 256) {
        wl[i] = coef_w[(size_t)co * CF * 9 + i];
        wf[i] = freq_w[(size_t)co * CF * 9 + i];
    }
    __syncthreads();
    const int pix = blockIdx.x * 256 + tid;
    const int y = pix / W_, x = pix % W_;
    float accC = coef_b[co], accF = freq_b[co];
    for (int ci = 0; ci < CF; ++ci) {
        const float* fp = feat + ((size_t)(n * CF + ci)) * HW;
        #pragma unroll
        for (int dy = -1; dy <= 1; ++dy) {
            const int yy = y + dy;
            if (yy < 0 || yy >= H_) continue;
            #pragma unroll
            for (int dx = -1; dx <= 1; ++dx) {
                const int xx = x + dx;
                if (xx < 0 || xx >= W_) continue;
                const float f  = fp[yy * W_ + xx];
                const int   wi = ci * 9 + (dy + 1) * 3 + (dx + 1);
                accC = fmaf(wl[wi], f, accC);
                accF = fmaf(wf[wi], f, accF);
            }
        }
    }
    __bf16* cp = side ? coefR : coefL;
    __bf16* fq = side ? freqR : freqL;
    cp[((size_t)(n * HID + co)) * HW + pix] = (__bf16)accC;
    fq[((size_t)(n * HID + co)) * HW + pix] = (__bf16)accF;
}

// ================= weight pre-pack into WMMA B-fragment tiles =================
__global__ __launch_bounds__(256) void pack_kernel(
    const float* __restrict__ w1, const float* __restrict__ w2,
    const float* __restrict__ w3, const float* __restrict__ w4,
    const float* __restrict__ w5,
    const float* __restrict__ b1, const float* __restrict__ b2,
    const float* __restrict__ b3, const float* __restrict__ b4,
    const float* __restrict__ b5,
    __bf16* __restrict__ wpk, float* __restrict__ bpk)
{
    const int idx = blockIdx.x * 256 + threadIdx.x;
    if (idx < WTOT) {
        const float* w; int K, base, rowsA;
        if      (idx < WO2) { w = w1; K = 128; base = WO1; rowsA = 256; }
        else if (idx < WO3) { w = w2; K = 256; base = WO2; rowsA = 256; }
        else if (idx < WO4) { w = w3; K = 256; base = WO3; rowsA = 256; }
        else if (idx < WO5) { w = w4; K = 256; base = WO4; rowsA = 256; }
        else                { w = w5; K = 256; base = WO5; rowsA = 3;   }
        const int KT   = K >> 5;
        const int e    = idx - base;
        const int tile = e >> 9;          // 512 elems per 32x16 tile
        const int rrr  = e & 511;
        const int lane = rrr >> 4;
        const int V    = (rrr >> 1) & 7;
        const int h    = rrr & 1;
        const int kt   = tile % KT;
        const int nt   = tile / KT;
        const int ncol = nt * 16 + (lane & 15);
        const int k    = kt * 32 + frag_kbase(V, lane) + h;
        const float v  = (ncol < rowsA) ? w[(size_t)ncol * K + k] : 0.f;
        wpk[idx] = (__bf16)v;
    } else if (idx < WTOT + 1040) {
        const int j = idx - WTOT;
        float v;
        if      (j < 256)  v = b1[j];
        else if (j < 512)  v = b2[j - 256];
        else if (j < 768)  v = b3[j - 512];
        else if (j < 1024) v = b4[j - 768];
        else { const int jj = j - 1024; v = (jj < 3) ? b5[jj] : 0.f; }
        bpk[j] = v;
    }
}

// ================= one MLP layer on a wave-private 16-row tile =================
template <int K, int N, bool RELU>
DI void layer_wmma(const __bf16* __restrict__ bufIn, __bf16* __restrict__ bufOut,
                   const __bf16* __restrict__ wp, const float* __restrict__ bp,
                   int rowBase, int lane)
{
    constexpr int KT = K / 32;
    constexpr int NT = N / 16;
    const int m  = lane & 15;
    const int g8 = (lane >> 4) * 8;

    union AF { v16bf v; unsigned u[8]; };
    AF af[KT];
    #pragma unroll
    for (int kt = 0; kt < KT; ++kt) {
        #pragma unroll
        for (int V = 0; V < 8; ++V) {
            const int kb = frag_kbase(V, lane);
            af[kt].u[V] = *(const unsigned*)(&bufIn[(rowBase + m) * STRIDE + kt * 32 + kb]);
        }
    }

    #pragma unroll 2
    for (int nt = 0; nt < NT; ++nt) {
        v8f acc = {0.f, 0.f, 0.f, 0.f, 0.f, 0.f, 0.f, 0.f};
        const __bf16* wt = wp + ((size_t)nt * KT) * 512 + lane * 16;
        #pragma unroll
        for (int kt = 0; kt < KT; ++kt) {
            union BF { v16bf v; u32x4 q[2]; } bf;
            const u32x4* p = (const u32x4*)(wt + (size_t)kt * 512);
            bf.q[0] = p[0];
            bf.q[1] = p[1];
            if (kt + 1 < KT) __builtin_prefetch(wt + (size_t)(kt + 1) * 512, 0, 1);
            acc = __builtin_amdgcn_wmma_f32_16x16x32_bf16(
                false, af[kt].v, false, bf.v, (short)0, acc, false, false);
        }
        const int   ncol = nt * 16 + m;
        const float bias = bp[ncol];
        #pragma unroll
        for (int r = 0; r < 8; ++r) {
            float y = acc[r] + bias;
            if (RELU) y = fmaxf(y, 0.f);
            bufOut[(rowBase + r + g8) * STRIDE + ncol] = (__bf16)y;
        }
    }
}

// ================= fused query kernel =================
__global__ __launch_bounds__(256) void query_kernel(
    const float* __restrict__ inp_l, const float* __restrict__ inp_r,
    const float* __restrict__ coord, const float* __restrict__ cell,
    const float* __restrict__ phase_w,
    const __bf16* __restrict__ coefL, const __bf16* __restrict__ coefR,
    const __bf16* __restrict__ freqL, const __bf16* __restrict__ freqR,
    const __bf16* __restrict__ wpk, const float* __restrict__ bpk,
    float* __restrict__ out)
{
    __shared__ __bf16 bufA[ROWS * STRIDE];
    __shared__ __bf16 bufB[ROWS * STRIDE];
    __shared__ float  areaS[ROWS];

    const int tid   = threadIdx.x;
    const int qtile = blockIdx.x;
    const int n     = blockIdx.y;
    const int side  = blockIdx.z;
    const __bf16* coefT = side ? coefR : coefL;
    const __bf16* freqT = side ? freqR : freqL;
    const float*  inp   = side ? inp_r : inp_l;

    // ---- prep: build X[128 rows x 128] in bufA, record areas ----
    {
        const int r  = tid >> 1;
        const int c0 = (tid & 1) * 64;
        const int ql = r >> 2;
        const int j  = r & 3;                 // (vx,vy): 0:(-,-) 1:(-,+) 2:(+,-) 3:(+,+)
        const int q  = qtile * QT + ql;
        const float vx = (j & 2) ? 1.f : -1.f;
        const float vy = (j & 1) ? 1.f : -1.f;
        const float cy = coord[((size_t)n * Q_ + q) * 2 + 0];
        const float cx = coord[((size_t)n * Q_ + q) * 2 + 1];
        float yc = cy + vx * (1.f / H_) + 1e-6f;
        float xc = cx + vy * (1.f / W_) + 1e-6f;
        yc = fminf(fmaxf(yc, -1.f + 1e-6f), 1.f - 1e-6f);
        xc = fminf(fmaxf(xc, -1.f + 1e-6f), 1.f - 1e-6f);
        const int iy = (int)fminf(fmaxf(rintf(((yc + 1.f) * H_ - 1.f) * 0.5f), 0.f), (float)(H_ - 1));
        const int ix = (int)fminf(fmaxf(rintf(((xc + 1.f) * W_ - 1.f) * 0.5f), 0.f), (float)(W_ - 1));
        const float qy = -1.f + (2.f * iy + 1.f) / H_;
        const float qx = -1.f + (2.f * ix + 1.f) / W_;
        const float rel0 = (cy - qy) * H_;
        const float rel1 = (cx - qx) * W_;
        if ((tid & 1) == 0) areaS[r] = fabsf(rel0 * rel1) + 1e-9f;
        const float rc0 = cell[((size_t)n * Q_ + q) * 2 + 0] * H_;
        const float rc1 = cell[((size_t)n * Q_ + q) * 2 + 1] * W_;
        const size_t gbase = ((size_t)n * HID) * HW + (size_t)(iy * W_ + ix);
        #pragma unroll 4
        for (int cc = 0; cc < 64; ++cc) {
            const int   c   = c0 + cc;
            const float cf  = (float)coefT[gbase + (size_t)c * HW];
            const float fr  = (float)freqT[gbase + (size_t)c * HW];
            const float rel = (c & 64) ? rel1 : rel0;
            const int   b   = c & 31;
            const float ang = rel * (3.14159265358979f * (float)(1u << b));
            const float pe  = (c & 32) ? __cosf(ang) : __sinf(ang);
            const float ph  = rc0 * phase_w[c * 2 + 0] + rc1 * phase_w[c * 2 + 1];
            bufA[r * STRIDE + c] = (__bf16)(cf * (fr * pe + ph));
        }
    }
    __syncthreads();

    // ---- 5-layer MLP; each wave owns a private 16-row tile -> no barriers ----
    const int lane    = tid & 31;
    const int rowBase = (tid >> 5) * 16;
    layer_wmma<128, 256, true >(bufA, bufB, wpk + WO1, bpk +    0, rowBase, lane);
    layer_wmma<256, 256, true >(bufB, bufA, wpk + WO2, bpk +  256, rowBase, lane);
    layer_wmma<256, 256, true >(bufA, bufB, wpk + WO3, bpk +  512, rowBase, lane);
    layer_wmma<256, 256, true >(bufB, bufA, wpk + WO4, bpk +  768, rowBase, lane);
    layer_wmma<256,  16, false>(bufA, bufB, wpk + WO5, bpk + 1024, rowBase, lane);
    __syncthreads();

    // ---- combine 4 neighbor preds (area-swapped) + bilinear-border lr ----
    if (tid < QT * 3) {
        const int ql = tid / 3, ch = tid % 3;
        const int q  = qtile * QT + ql;
        const float a0 = areaS[4 * ql + 0], a1 = areaS[4 * ql + 1];
        const float a2 = areaS[4 * ql + 2], a3 = areaS[4 * ql + 3];
        const float inv = 1.f / (a0 + a1 + a2 + a3);
        const float wgt[4] = {a3 * inv, a2 * inv, a1 * inv, a0 * inv};
        float ret = 0.f;
        #pragma unroll
        for (int j = 0; j < 4; ++j)
            ret += (float)bufB[(4 * ql + j) * STRIDE + ch] * wgt[j];
        const float cy = coord[((size_t)n * Q_ + q) * 2 + 0];
        const float cx = coord[((size_t)n * Q_ + q) * 2 + 1];
        const float fy = ((cy + 1.f) * H_ - 1.f) * 0.5f;
        const float fx = ((cx + 1.f) * W_ - 1.f) * 0.5f;
        const float y0 = floorf(fy), x0 = floorf(fx);
        const float wy = fy - y0,    wx = fx - x0;
        const int y0c = (int)fminf(fmaxf(y0,       0.f), (float)(H_ - 1));
        const int y1c = (int)fminf(fmaxf(y0 + 1.f, 0.f), (float)(H_ - 1));
        const int x0c = (int)fminf(fmaxf(x0,       0.f), (float)(W_ - 1));
        const int x1c = (int)fminf(fmaxf(x0 + 1.f, 0.f), (float)(W_ - 1));
        const float* ip = inp + ((size_t)n * 3 + ch) * HW;
        const float bil = ip[y0c * W_ + x0c] * (1.f - wy) * (1.f - wx)
                        + ip[y0c * W_ + x1c] * (1.f - wy) * wx
                        + ip[y1c * W_ + x0c] * wy * (1.f - wx)
                        + ip[y1c * W_ + x1c] * wy * wx;
        out[((size_t)n * Q_ + q) * 6 + side * 3 + ch] = ret + bil;
    }
}

// ================= host launch =================
extern "C" void kernel_launch(void* const* d_in, const int* in_sizes, int n_in,
                              void* d_out, int out_size, void* d_ws, size_t ws_size,
                              hipStream_t stream) {
    const float* inp_l   = (const float*)d_in[0];
    const float* inp_r   = (const float*)d_in[1];
    const float* feat_l  = (const float*)d_in[2];
    const float* feat_r  = (const float*)d_in[3];
    const float* coord   = (const float*)d_in[4];
    const float* cell    = (const float*)d_in[5];
    const float* coef_w  = (const float*)d_in[6];
    const float* coef_b  = (const float*)d_in[7];
    const float* freq_w  = (const float*)d_in[8];
    const float* freq_b  = (const float*)d_in[9];
    const float* phase_w = (const float*)d_in[10];
    const float* w1 = (const float*)d_in[11]; const float* b1 = (const float*)d_in[12];
    const float* w2 = (const float*)d_in[13]; const float* b2 = (const float*)d_in[14];
    const float* w3 = (const float*)d_in[15]; const float* b3 = (const float*)d_in[16];
    const float* w4 = (const float*)d_in[17]; const float* b4 = (const float*)d_in[18];
    const float* w5 = (const float*)d_in[19]; const float* b5 = (const float*)d_in[20];

    char* ws = (char*)d_ws;
    __bf16* coefL = (__bf16*)(ws + OFF_COEF0);
    __bf16* coefR = (__bf16*)(ws + OFF_COEF1);
    __bf16* freqL = (__bf16*)(ws + OFF_FREQ0);
    __bf16* freqR = (__bf16*)(ws + OFF_FREQ1);
    __bf16* wpk   = (__bf16*)(ws + OFF_WPK);
    float*  bpk   = (float*) (ws + OFF_BPK);
    float*  outp  = (float*)d_out;

    conv3x3_kernel<<<dim3(HW / 256, HID, 4), 256, 0, stream>>>(
        feat_l, feat_r, coef_w, coef_b, freq_w, freq_b, coefL, coefR, freqL, freqR);

    pack_kernel<<<dim3((WTOT + 1040 + 255) / 256), 256, 0, stream>>>(
        w1, w2, w3, w4, w5, b1, b2, b3, b4, b5, wpk, bpk);

    query_kernel<<<dim3(Q_ / QT, N_, 2), 256, 0, stream>>>(
        inp_l, inp_r, coord, cell, phase_w, coefL, coefR, freqL, freqR, wpk, bpk, outp);
}
